// MEGDecoder_54674933678309
// MI455X (gfx1250) — matrix operand order repaired and checked by reading, not compiled, not run on hardware
//
// MEG decoder for MI455X (gfx1250).
// All GEMMs use v_wmma_f32_16x16x32_f16 with operands pre-packed to f16 in
// WMMA-native layouts (A: row-major [M][Kp], B: n-major [N][Kp], Kp % 32 == 0)
// inside a once-zeroed f16 arena: GEMM inner loops are pure clause'd b128
// loads + wmma. Large GEMMs use M-pair register blocking (2 C-tiles/wave
// sharing the B fragment). LSTM recurrence: one 32-wave workgroup on a single
// WGP, h in LDS (f16), B fragment shared across all 4 gate WMMAs per k-step.
#include <hip/hip_runtime.h>
#include <hip/hip_bf16.h>

typedef __attribute__((ext_vector_type(16))) _Float16 v16h;
typedef __attribute__((ext_vector_type(8)))  _Float16 v8h;
typedef __attribute__((ext_vector_type(8)))  float    v8f;

#define DEV static __device__ __forceinline__

DEV v8f wmma16(const v16h a, const v16h b, const v8f c) {
  return __builtin_amdgcn_wmma_f32_16x16x32_f16(false, a, false, b, (short)0, c,
                                                false, false);
}
DEV float sigf(float x) { return 1.f / (1.f + __expf(-x)); }
DEV v16h cat8(const v8h lo, const v8h hi) {
  return __builtin_shufflevector(lo, hi, 0, 1, 2, 3, 4, 5, 6, 7, 8, 9, 10, 11,
                                 12, 13, 14, 15);
}

// ---- single 16x16 C tile ----
// A frag: lane<16 -> row m0+lane, K(k0+0..15); lane>=16 K(k0+16..31).
// B frag: lane<16 -> col, K(k0+0..15); lane>=16 K(k0+16..31).
DEV v8f gemm_tile(const _Float16* __restrict__ A, const _Float16* __restrict__ Bn,
                  int Kp, int m0, int col, int kbeg, int kend) {
  const int lane = threadIdx.x & 31;
  const _Float16* ap = A + (long)(m0 + (lane & 15)) * Kp + ((lane & 16) ? 16 : 0);
  const _Float16* bp = Bn + (long)col * Kp + ((lane & 16) ? 16 : 0);
  v8f acc;
#pragma unroll
  for (int i = 0; i < 8; ++i) acc[i] = 0.f;
  for (int k0 = kbeg; k0 < kend; k0 += 32) {
    v8h alo = *(const v8h*)(ap + k0);
    v8h ahi = *(const v8h*)(ap + k0 + 8);
    v16h b  = *(const v16h*)(bp + k0);
    acc = wmma16(cat8(alo, ahi), b, acc);
  }
  return acc;
}

// ---- M-pair: two adjacent 16-row C tiles sharing one B fragment ----
DEV void gemm_tile_pair(const _Float16* __restrict__ A,
                        const _Float16* __restrict__ Bn, int Kp, int m0,
                        int col, v8f& c0, v8f& c1) {
  const int lane = threadIdx.x & 31;
  const _Float16* ap0 = A + (long)(m0 + (lane & 15)) * Kp + ((lane & 16) ? 16 : 0);
  const _Float16* ap1 = ap0 + (long)16 * Kp;
  const _Float16* bp = Bn + (long)col * Kp + ((lane & 16) ? 16 : 0);
#pragma unroll
  for (int i = 0; i < 8; ++i) { c0[i] = 0.f; c1[i] = 0.f; }
  for (int k0 = 0; k0 < Kp; k0 += 32) {
    v16h b = *(const v16h*)(bp + k0);
    v8h lo0 = *(const v8h*)(ap0 + k0);
    v8h hi0 = *(const v8h*)(ap0 + k0 + 8);
    v8h lo1 = *(const v8h*)(ap1 + k0);
    v8h hi1 = *(const v8h*)(ap1 + k0 + 8);
    c0 = wmma16(cat8(lo0, hi0), b, c0);
    c1 = wmma16(cat8(lo1, hi1), b, c1);
  }
}
// C/D frag: vgpr i -> row m0 + i + (lane>=16 ? 8 : 0), col n0 + (lane&15)

// ===================== pack / utility kernels =====================
__global__ __launch_bounds__(256) void zero16_kernel(uint4* __restrict__ p, long n16) {
  long i = (long)blockIdx.x * 256 + threadIdx.x;
  if (i < n16) p[i] = uint4{0u, 0u, 0u, 0u};
}
__global__ __launch_bounds__(256) void pack_f16_kernel(
    const float* __restrict__ src, _Float16* __restrict__ dst, long n) {
  long i = (long)blockIdx.x * 256 + threadIdx.x;
  if (i < n) dst[i] = (_Float16)src[i];
}
// f32 [M][K] -> f16 [M][Kp] (pads stay zero from arena clear)
__global__ __launch_bounds__(256) void pack_w_kernel(
    const float* __restrict__ src, _Float16* __restrict__ dst, int M, int K, int Kp) {
  long i = (long)blockIdx.x * 256 + threadIdx.x;
  if (i >= (long)M * K) return;
  int m = (int)(i / K), k = (int)(i - (long)m * K);
  dst[(long)m * Kp + k] = (_Float16)src[i];
}
// rel_emb [101][64] -> relT f16 [64][128] (transposed, pads zero)
__global__ __launch_bounds__(256) void pack_relT_kernel(
    const float* __restrict__ rel, _Float16* __restrict__ relT) {
  int i = blockIdx.x * 256 + threadIdx.x;
  if (i >= 101 * 64) return;
  int j = i >> 6, c = i & 63;
  relT[(long)c * 128 + j] = (_Float16)rel[i];
}
// meg [8][273][2048] -> im2col f16 [8208][1120] (k=4,s=2,p=2)
__global__ __launch_bounds__(256) void pack_im2col1_kernel(
    const float* __restrict__ X, _Float16* __restrict__ C) {
  long i = (long)blockIdx.x * 256 + threadIdx.x;
  if (i >= 8200L * 1092) return;
  int n = (int)(i / 1092), kk = (int)(i - (long)n * 1092);
  int b = n / 1025, l = n - b * 1025;
  int ci = kk >> 2, kw = kk & 3;
  int tp = 2 * l + kw - 2;
  float v = (tp >= 0 && tp < 2048) ? X[((long)b * 273 + ci) * 2048 + tp] : 0.f;
  C[(long)n * 1120 + kk] = (_Float16)v;
}
// Y1 f32 [8][512][1025] -> im2col f16 [4112][2048]
__global__ __launch_bounds__(256) void pack_im2col2_kernel(
    const float* __restrict__ X, _Float16* __restrict__ C) {
  long i = (long)blockIdx.x * 256 + threadIdx.x;
  if (i >= 4104L * 2048) return;
  int n = (int)(i >> 11), kk = (int)(i & 2047);
  int b = n / 513, l = n - b * 513;
  int ci = kk >> 2, kw = kk & 3;
  int tp = 2 * l + kw - 2;
  float v = (tp >= 0 && tp < 1025) ? X[((long)b * 512 + ci) * 1025 + tp] : 0.f;
  C[(long)n * 2048 + kk] = (_Float16)v;
}
// LSTM0 input: [conv feats(512) | subj_emb(64)] f16 [4112][576], n = t*8+b
__global__ __launch_bounds__(256) void pack_x0_kernel(
    const float* __restrict__ Y2, const float* __restrict__ subj_emb,
    const int* __restrict__ subjects, _Float16* __restrict__ C) {
  long i = (long)blockIdx.x * 256 + threadIdx.x;
  if (i >= 4104L * 576) return;
  int n = (int)(i / 576), c = (int)(i - (long)n * 576);
  int b = n & 7, t = n >> 3;
  float v = (c < 512) ? Y2[((long)b * 512 + c) * 513 + t]
                      : subj_emb[(long)subjects[b] * 64 + (c - 512)];
  C[(long)n * 576 + c] = (_Float16)v;
}

// ===================== GEMM kernels =====================
// Conv as GEMM (M-pair blocked): A=W f16 [Cout][Kp], B=im2col f16,
// out f32 [B][Cout][Lout] + bias + ReLU
__global__ __launch_bounds__(256) void conv_gemm_kernel(
    const _Float16* __restrict__ W, const float* __restrict__ bias,
    const _Float16* __restrict__ Xc, float* __restrict__ Y,
    int Cout, int Kp, int Lout) {
  const int NN = 8 * Lout;
  const int tiles_m2 = Cout >> 5, tiles_n = (NN + 15) >> 4;
  const int wid = (blockIdx.x << 3) + (threadIdx.x >> 5);
  if (wid >= tiles_m2 * tiles_n) return;
  const int tm = wid % tiles_m2, tn = wid / tiles_m2;
  const int m0 = tm << 5;
  const int lane = threadIdx.x & 31;
  const int col = (tn << 4) + (lane & 15);
  v8f a0, a1;
  gemm_tile_pair(W, Xc, Kp, m0, col, a0, a1);
  if (col >= NN) return;
  const int bb = col / Lout, ll = col - bb * Lout;
  const int rb = m0 + ((lane & 16) ? 8 : 0);
#pragma unroll
  for (int i = 0; i < 8; ++i) {
    int r0 = rb + i, r1 = rb + 16 + i;
    Y[((long)bb * Cout + r0) * Lout + ll] = fmaxf(a0[i] + bias[r0], 0.f);
    Y[((long)bb * Cout + r1) * Lout + ll] = fmaxf(a1[i] + bias[r1], 0.f);
  }
}

// Generic n-major GEMM (M-pair blocked): Out f32 [n][Cout] = Bn.A^T + biases
__global__ __launch_bounds__(256) void gemm_n_kernel(
    const _Float16* __restrict__ A, const _Float16* __restrict__ Bn,
    const float* __restrict__ bias, const float* __restrict__ bias2,
    float* __restrict__ Out, int Cout, int Kp) {
  const int tiles_m2 = Cout >> 5, tiles_n = 257;
  const int wid = (blockIdx.x << 3) + (threadIdx.x >> 5);
  if (wid >= tiles_m2 * tiles_n) return;
  const int tm = wid % tiles_m2, tn = wid / tiles_m2;
  const int m0 = tm << 5;
  const int lane = threadIdx.x & 31;
  const int col = (tn << 4) + (lane & 15);
  v8f a0, a1;
  gemm_tile_pair(A, Bn, Kp, m0, col, a0, a1);
  if (col >= 4104) return;
  const int rb = m0 + ((lane & 16) ? 8 : 0);
#pragma unroll
  for (int i = 0; i < 8; ++i) {
    int r0 = rb + i, r1 = rb + 16 + i;
    float v0 = a0[i] + bias[r0] + (bias2 ? bias2[r0] : 0.f);
    float v1 = a1[i] + bias[r1] + (bias2 ? bias2[r1] : 0.f);
    Out[(long)col * Cout + r0] = v0;
    Out[(long)col * Cout + r1] = v1;
  }
}

// q/k/cnt/out projections from H f16 [4112][512] (M-pair blocked).
// mode 0: f16 [bh][528][64] (A-layout for QK/P GEMMs; q and k)
// mode 1: f16 [bh][64][544] (A-layout for AV GEMM; cnt)
// mode 3: f32 d_out [b][128][513]
__global__ __launch_bounds__(256) void proj_gemm_kernel(
    const _Float16* __restrict__ A, const float* __restrict__ bias,
    const _Float16* __restrict__ Bn, void* __restrict__ Out, int Cout, int mode) {
  const int tiles_m2 = Cout >> 5, tiles_n = 257;
  const int wid = (blockIdx.x << 3) + (threadIdx.x >> 5);
  if (wid >= tiles_m2 * tiles_n) return;
  const int tm = wid % tiles_m2, tn = wid / tiles_m2;
  const int m0 = tm << 5;
  const int lane = threadIdx.x & 31;
  const int col = (tn << 4) + (lane & 15);
  v8f a0, a1;
  gemm_tile_pair(A, Bn, 512, m0, col, a0, a1);
  if (col >= 4104) return;
  const int rb = m0 + ((lane & 16) ? 8 : 0);
  const int b = col & 7, t = col >> 3;
#pragma unroll
  for (int u = 0; u < 16; ++u) {
    int row = rb + (u & 7) + ((u & 8) ? 16 : 0);
    float v = ((u & 8) ? a1[u & 7] : a0[u & 7]) + bias[row];
    if (mode == 0) {
      int h = row >> 6, d = row & 63;
      ((_Float16*)Out)[((long)(b * 8 + h) * 528 + t) * 64 + d] = (_Float16)v;
    } else if (mode == 1) {
      int h = row >> 6, d = row & 63;
      ((_Float16*)Out)[((long)(b * 8 + h) * 64 + d) * 544 + t] = (_Float16)v;
    } else {
      ((float*)Out)[((long)b * 128 + row) * 513 + t] = v;
    }
  }
}

// QK^T per (b,h), band-skipped tiles. A=Qf16 [528][64], B=Kf16 [528][64].
__global__ __launch_bounds__(256) void qk_dots_kernel(
    const _Float16* __restrict__ Qf, const _Float16* __restrict__ Kf,
    float* __restrict__ DOTS) {
  const int bh = blockIdx.y;
  const int wid = (blockIdx.x << 3) + (threadIdx.x >> 5);
  if (wid >= 33 * 33) return;
  const int tm = wid % 33, tn = wid / 33;
  const int t0 = tm << 4, s0 = tn << 4;
  if (t0 > s0 + 65 || s0 > t0 + 65) return;
  const int lane = threadIdx.x & 31;
  const int col = s0 + (lane & 15);
  v8f acc = gemm_tile(Qf + (long)bh * 528 * 64, Kf + (long)bh * 528 * 64, 64,
                      t0, col, 0, 64);
  if (col >= 513) return;
  const int rb = t0 + ((lane & 16) ? 8 : 0);
#pragma unroll
  for (int i = 0; i < 8; ++i) {
    int row = rb + i;
    if (row < 513) DOTS[(long)bh * 263169 + (long)row * 513 + col] = acc[i];
  }
}

// P[bh][t][jr] = q[:,t].rel_emb[jr]. B = relB f16 [112][64].
__global__ __launch_bounds__(256) void prel_kernel(
    const _Float16* __restrict__ Qf, const _Float16* __restrict__ relB,
    float* __restrict__ P) {
  const int bh = blockIdx.y;
  const int wid = (blockIdx.x << 3) + (threadIdx.x >> 5);
  if (wid >= 33 * 7) return;
  const int tm = wid % 33, tn = wid / 33;
  const int t0 = tm << 4;
  const int lane = threadIdx.x & 31;
  const int col = (tn << 4) + (lane & 15);  // jr < 112
  v8f acc = gemm_tile(Qf + (long)bh * 528 * 64, relB, 64, t0, col, 0, 64);
  const int rb = t0 + ((lane & 16) ? 8 : 0);
#pragma unroll
  for (int i = 0; i < 8; ++i) {
    int row = rb + i;
    if (row < 513) P[(long)bh * 513 * 112 + (long)row * 112 + col] = acc[i];
  }
}

// banded softmax, one wave per (bh,t). Emits f16 WT row [t][544] (zeros
// outside band) and f16 WR row [t][128] (band-compacted, jr = 50+t-s).
__global__ __launch_bounds__(256) void softmax_kernel(
    const float* __restrict__ DOTS, const float* __restrict__ P,
    _Float16* __restrict__ WT, _Float16* __restrict__ WR) {
  const int gid = blockIdx.x * 8 + (threadIdx.x >> 5);
  if (gid >= 64 * 513) return;
  const int bh = gid / 513, t = gid % 513;
  const int lane = threadIdx.x & 31;
  const float* drow = DOTS + (long)bh * 263169 + (long)t * 513;
  const float* prow = P + (long)bh * 513 * 112 + (long)t * 112;
  _Float16* wtrow = WT + (long)bh * 528 * 544 + (long)t * 544;
  _Float16* wrrow = WR + (long)bh * 528 * 128 + (long)t * 128;
  for (int j = lane; j < 544; j += 32) wtrow[j] = (_Float16)0.f;
  for (int j = lane; j < 128; j += 32) wrrow[j] = (_Float16)0.f;
  int lo = t - 50; if (lo < 0) lo = 0;
  int hb = t + 50; if (hb > 512) hb = 512;
  float vmax = -3.0e38f, vals[4];
#pragma unroll
  for (int it = 0; it < 4; ++it) {
    int s = lo + it * 32 + lane;
    float v = -3.0e38f;
    if (s <= hb) v = drow[s] + 0.3f * prow[50 + t - s];
    vals[it] = v;
    vmax = fmaxf(vmax, v);
  }
#pragma unroll
  for (int off = 16; off; off >>= 1) vmax = fmaxf(vmax, __shfl_xor(vmax, off, 32));
  float sum = 0.f;
#pragma unroll
  for (int it = 0; it < 4; ++it) {
    float e = (vals[it] > -1.0e37f) ? __expf(vals[it] - vmax) : 0.f;
    vals[it] = e;
    sum += e;
  }
#pragma unroll
  for (int off = 16; off; off >>= 1) sum += __shfl_xor(sum, off, 32);
  float inv = 1.f / sum;
#pragma unroll
  for (int it = 0; it < 4; ++it) {
    int s = lo + it * 32 + lane;
    if (s <= hb) {
      float w = vals[it] * inv;
      wtrow[s] = (_Float16)w;
      wrrow[50 + t - s] = (_Float16)w;
    }
  }
}

// O1[c,t] = sum_s cnt[c,s]*w[t,s]. A=CNTf16 [64][544], B=WTf16 [528][544].
// K-loop restricted to band tiles (WT rows are zero outside the band).
__global__ __launch_bounds__(256) void av_gemm_kernel(
    const _Float16* __restrict__ CNTf, const _Float16* __restrict__ WTf,
    float* __restrict__ ATT) {
  const int bh = blockIdx.y;
  const int b = bh >> 3, h = bh & 7;
  const int wid = (blockIdx.x << 3) + (threadIdx.x >> 5);
  if (wid >= 4 * 33) return;
  const int tm = wid & 3, tn = wid >> 2;
  const int m0 = tm << 4, n0 = tn << 4;
  int klo = n0 - 50; if (klo < 0) klo = 0; klo &= ~31;
  int khi = n0 + 66; if (khi > 544) khi = 544;
  const int lane = threadIdx.x & 31;
  const int col = n0 + (lane & 15);  // t
  v8f acc = gemm_tile(CNTf + (long)bh * 64 * 544, WTf + (long)bh * 528 * 544,
                      544, m0, col, klo, khi);
  if (col >= 513) return;
  const int rb = m0 + ((lane & 16) ? 8 : 0);
#pragma unroll
  for (int i = 0; i < 8; ++i) {
    int c = rb + i;
    ATT[((long)col * 8 + b) * 512 + (h << 6) + c] = acc[i];
  }
}

// ATT += 0.3 * sum_jr WR[t,jr] * rel_emb[jr,c]. A=WRf16 [528][128], B=relT [64][128].
__global__ __launch_bounds__(256) void relout_gemm_kernel(
    const _Float16* __restrict__ WRf, const _Float16* __restrict__ relT,
    float* __restrict__ ATT) {
  const int bh = blockIdx.y;
  const int b = bh >> 3, h = bh & 7;
  const int wid = (blockIdx.x << 3) + (threadIdx.x >> 5);
  if (wid >= 33 * 4) return;
  const int tm = wid % 33, tn = wid / 33;
  const int m0 = tm << 4;
  const int lane = threadIdx.x & 31;
  const int col = (tn << 4) + (lane & 15);  // c < 64
  v8f acc = gemm_tile(WRf + (long)bh * 528 * 128, relT, 128, m0, col, 0, 128);
  const int rb = m0 + ((lane & 16) ? 8 : 0);
#pragma unroll
  for (int i = 0; i < 8; ++i) {
    int t = rb + i;
    if (t < 513) {
      float* p = ATT + ((long)t * 8 + b) * 512 + (h << 6) + col;
      *p += 0.3f * acc[i];
    }
  }
}

// ===================== LSTM recurrence (single WGP) =====================
// Wave w owns hidden slice [16w,16w+16); per k-step the h fragment is loaded
// from LDS ONCE and fed to 4 independent WMMAs (i/f/g/o accumulators).
__global__ __launch_bounds__(1024, 1) void lstm_rec_kernel(
    const _Float16* __restrict__ Whh16,  // [2048][512] f16
    const float* __restrict__ G,         // [T*8][2048] precomputed gates
    float* __restrict__ Hout,            // [T*8][512] f32 (residual use)
    _Float16* __restrict__ H16,          // [4112][512] f16 n-major
    int T) {
  __shared__ alignas(64) _Float16 hls[16 * 512];  // h: [b(16)][c(512)]
  const int tid = threadIdx.x, wave = tid >> 5, lane = tid & 31;
  for (int i = tid; i < (16 * 512) / 2; i += 1024) ((unsigned int*)hls)[i] = 0u;
  __syncthreads();
  const int jh0 = wave << 4;
  const int col = lane & 15;  // batch (valid < 8)
  const int hi = (lane & 16) ? 8 : 0;
  const int offA = (lane & 16) ? 16 : 0;
  const int offB = (lane & 16) ? 16 : 0;
  const bool bvalid = col < 8;
  const _Float16* apg[4];
#pragma unroll
  for (int g = 0; g < 4; ++g)
    apg[g] = Whh16 + (long)((g << 9) + jh0 + (lane & 15)) * 512 + offA;
  const _Float16* bp = hls + (col << 9) + offB;
  float cst[8];
#pragma unroll
  for (int i = 0; i < 8; ++i) cst[i] = 0.f;

  for (int t = 0; t < T; ++t) {
    if (bvalid && t + 1 < T)
      __builtin_prefetch(G + (long)((t + 1) * 8 + col) * 2048 + jh0, 0, 0);
    v8f acc[4];
#pragma unroll
    for (int g = 0; g < 4; ++g) {
      if (bvalid) {
        const float4* gp =
            (const float4*)(G + (long)(t * 8 + col) * 2048 + (g << 9) + jh0 + hi);
        float4 ga = gp[0], gb = gp[1];
        acc[g][0] = ga.x; acc[g][1] = ga.y; acc[g][2] = ga.z; acc[g][3] = ga.w;
        acc[g][4] = gb.x; acc[g][5] = gb.y; acc[g][6] = gb.z; acc[g][7] = gb.w;
      } else {
#pragma unroll
        for (int i = 0; i < 8; ++i) acc[g][i] = 0.f;
      }
    }
#pragma unroll 4
    for (int ks = 0; ks < 16; ++ks) {
      const int k0 = ks << 5;
      v16h b = *(const v16h*)(bp + k0);  // one LDS frag, 4 WMMAs
#pragma unroll
      for (int g = 0; g < 4; ++g) {
        v8h alo = *(const v8h*)(apg[g] + k0);
        v8h ahi = *(const v8h*)(apg[g] + k0 + 8);
        acc[g] = wmma16(cat8(alo, ahi), b, acc[g]);
      }
    }
    __syncthreads();  // everyone done reading h(t-1)
#pragma unroll
    for (int i = 0; i < 8; ++i) {
      float iv = acc[0][i], fv = acc[1][i], gv = acc[2][i], ov = acc[3][i];
      float cc = sigf(fv) * cst[i] + sigf(iv) * tanhf(gv);
      float hh = sigf(ov) * tanhf(cc);
      cst[i] = cc;
      if (bvalid) {
        int j = jh0 + hi + i;
        long n = (long)(t * 8 + col);
        Hout[n * 512 + j] = hh;
        H16[n * 512 + j] = (_Float16)hh;
        hls[(col << 9) + j] = (_Float16)hh;
      }
    }
    __syncthreads();  // h(t) visible before next step
  }
}

// ===================== BN / epilogue =====================
__global__ __launch_bounds__(256) void bn_reduce_kernel(
    const float* __restrict__ FC, float* __restrict__ stats, int NN) {
  const int ch = blockIdx.x * 8 + (threadIdx.x >> 5);
  if (ch >= 512) return;
  const int lane = threadIdx.x & 31;
  float s = 0.f, s2 = 0.f;
  for (int n = lane; n < NN; n += 32) {
    float v = FC[(long)n * 512 + ch];
    s += v; s2 += v * v;
  }
#pragma unroll
  for (int off = 16; off; off >>= 1) {
    s += __shfl_xor(s, off, 32);
    s2 += __shfl_xor(s2, off, 32);
  }
  if (lane == 0) {
    float mu = s / (float)NN;
    stats[ch] = mu;
    stats[512 + ch] = s2 / (float)NN - mu * mu;
  }
}
__global__ __launch_bounds__(256) void bn_apply_kernel(
    const float* __restrict__ FC, const float* __restrict__ H1,
    const float* __restrict__ stats, const float* __restrict__ g,
    const float* __restrict__ bb, const float* __restrict__ sc,
    float* __restrict__ XF, long total) {
  long idx = (long)blockIdx.x * 256 + threadIdx.x;
  if (idx >= total) return;
  int c = (int)(idx & 511);
  float mu = stats[c], var = stats[512 + c];
  float xh = (FC[idx] - mu) * rsqrtf(var + 1e-5f) * g[c] + bb[c];
  XF[idx] = H1[idx] + fmaxf(xh, 0.f) * sc[c];
}

// =====================================================================
static inline unsigned cdiv(long a, long b) { return (unsigned)((a + b - 1) / b); }

extern "C" void kernel_launch(void* const* d_in, const int* in_sizes, int n_in,
                              void* d_out, int out_size, void* d_ws,
                              size_t ws_size, hipStream_t stream) {
  (void)in_sizes; (void)n_in; (void)out_size; (void)ws_size;
  const float* meg      = (const float*)d_in[0];
  const float* conv1_w  = (const float*)d_in[1];
  const float* conv1_b  = (const float*)d_in[2];
  const float* conv2_w  = (const float*)d_in[3];
  const float* conv2_b  = (const float*)d_in[4];
  const float* subj_emb = (const float*)d_in[5];
  const float* W_ih0 = (const float*)d_in[6];
  const float* W_hh0 = (const float*)d_in[7];
  const float* b_ih0 = (const float*)d_in[8];
  const float* b_hh0 = (const float*)d_in[9];
  const float* W_ih1 = (const float*)d_in[10];
  const float* W_hh1 = (const float*)d_in[11];
  const float* b_ih1 = (const float*)d_in[12];
  const float* b_hh1 = (const float*)d_in[13];
  const float* q_w = (const float*)d_in[14];
  const float* q_b = (const float*)d_in[15];
  const float* k_w = (const float*)d_in[16];
  const float* k_b = (const float*)d_in[17];
  const float* c_w = (const float*)d_in[18];
  const float* c_b = (const float*)d_in[19];
  const float* rel_emb = (const float*)d_in[20];
  const float* fc_w = (const float*)d_in[21];
  const float* fc_b = (const float*)d_in[22];
  const float* bn_g = (const float*)d_in[23];
  const float* bn_b = (const float*)d_in[24];
  const float* attn_scale = (const float*)d_in[25];
  const float* out_w = (const float*)d_in[26];
  const float* out_b = (const float*)d_in[27];
  const int* subjects = (const int*)d_in[28];

  char* ws = (char*)d_ws;
  size_t o = 0;
  auto alloc = [&](size_t bytes) -> char* {
    char* p = ws + o;
    o += (bytes + 255) & ~(size_t)255;
    return p;
  };
  // ---- f32 regions (phase-reused) ----
  float* DOTS = (float*)alloc(64ull * 513 * 513 * 4);
  char* Greg = alloc(4104ull * 2048 * 4);  // gates; later P
  float* G = (float*)Greg;
  float* P = (float*)Greg;
  char* r1 = alloc(16809984);  // Y1; later ATT + FC
  float* Y1 = (float*)r1;
  float* ATT = (float*)r1;
  float* FC = (float*)(r1 + 8404992);
  char* r2 = alloc(8404992);  // Y2; later H0 f32 scratch; later XF
  float* Y2 = (float*)r2;
  float* H0f = (float*)r2;
  float* XF = (float*)r2;
  float* H1f = (float*)alloc(8404992);
  float* stats = (float*)alloc(4096);
  // ---- f16 arena (zeroed once; pads stay zero) ----
  size_t f16_begin = o;
  _Float16* cw1f  = (_Float16*)alloc(512ull * 1120 * 2);
  _Float16* cw2f  = (_Float16*)alloc(512ull * 2048 * 2);
  _Float16* wih0f = (_Float16*)alloc(2048ull * 576 * 2);
  _Float16* whh0f = (_Float16*)alloc(2048ull * 512 * 2);
  _Float16* wih1f = (_Float16*)alloc(2048ull * 512 * 2);
  _Float16* whh1f = (_Float16*)alloc(2048ull * 512 * 2);
  _Float16* qwf   = (_Float16*)alloc(512ull * 512 * 2);
  _Float16* kwf   = (_Float16*)alloc(512ull * 512 * 2);
  _Float16* cwf   = (_Float16*)alloc(512ull * 512 * 2);
  _Float16* fcwf  = (_Float16*)alloc(512ull * 512 * 2);
  _Float16* outwf = (_Float16*)alloc(128ull * 512 * 2);
  _Float16* relBf = (_Float16*)alloc(112ull * 64 * 2);
  _Float16* relTf = (_Float16*)alloc(64ull * 128 * 2);
  _Float16* Xc1   = (_Float16*)alloc(8208ull * 1120 * 2);
  _Float16* Xc2   = (_Float16*)alloc(4112ull * 2048 * 2);
  _Float16* Xl0   = (_Float16*)alloc(4112ull * 576 * 2);
  _Float16* H16_0 = (_Float16*)alloc(4112ull * 512 * 2);
  _Float16* H16_1 = (_Float16*)alloc(4112ull * 512 * 2);
  _Float16* Qf    = (_Float16*)alloc(64ull * 528 * 64 * 2);
  _Float16* Kf    = (_Float16*)alloc(64ull * 528 * 64 * 2);
  _Float16* CNTf  = (_Float16*)alloc(64ull * 64 * 544 * 2);
  _Float16* WTf   = (_Float16*)alloc(64ull * 528 * 544 * 2);
  _Float16* WRf   = (_Float16*)alloc(64ull * 528 * 128 * 2);
  _Float16* ATTf  = (_Float16*)alloc(4112ull * 512 * 2);
  _Float16* XFf   = (_Float16*)alloc(4112ull * 512 * 2);
  size_t f16_bytes = o - f16_begin;

  const dim3 blk(256);
  // clear f16 arena (zero K/N pads everywhere, never NaN)
  long n16 = (long)(f16_bytes >> 4);
  zero16_kernel<<<cdiv(n16, 256), blk, 0, stream>>>((uint4*)(ws + f16_begin), n16);

  // weight packs
  pack_w_kernel<<<cdiv(512L * 1092, 256), blk, 0, stream>>>(conv1_w, cw1f, 512, 1092, 1120);
  pack_w_kernel<<<cdiv(512L * 2048, 256), blk, 0, stream>>>(conv2_w, cw2f, 512, 2048, 2048);
  pack_w_kernel<<<cdiv(2048L * 576, 256), blk, 0, stream>>>(W_ih0, wih0f, 2048, 576, 576);
  pack_w_kernel<<<cdiv(2048L * 512, 256), blk, 0, stream>>>(W_hh0, whh0f, 2048, 512, 512);
  pack_w_kernel<<<cdiv(2048L * 512, 256), blk, 0, stream>>>(W_ih1, wih1f, 2048, 512, 512);
  pack_w_kernel<<<cdiv(2048L * 512, 256), blk, 0, stream>>>(W_hh1, whh1f, 2048, 512, 512);
  pack_w_kernel<<<cdiv(512L * 512, 256), blk, 0, stream>>>(q_w, qwf, 512, 512, 512);
  pack_w_kernel<<<cdiv(512L * 512, 256), blk, 0, stream>>>(k_w, kwf, 512, 512, 512);
  pack_w_kernel<<<cdiv(512L * 512, 256), blk, 0, stream>>>(c_w, cwf, 512, 512, 512);
  pack_w_kernel<<<cdiv(512L * 512, 256), blk, 0, stream>>>(fc_w, fcwf, 512, 512, 512);
  pack_w_kernel<<<cdiv(128L * 512, 256), blk, 0, stream>>>(out_w, outwf, 128, 512, 512);
  pack_f16_kernel<<<cdiv(101L * 64, 256), blk, 0, stream>>>(rel_emb, relBf, 101L * 64);
  pack_relT_kernel<<<cdiv(101L * 64, 256), blk, 0, stream>>>(rel_emb, relTf);

  // encoder convs
  pack_im2col1_kernel<<<cdiv(8200L * 1092, 256), blk, 0, stream>>>(meg, Xc1);
  conv_gemm_kernel<<<1026, blk, 0, stream>>>(cw1f, conv1_b, Xc1, Y1, 512, 1120, 1025);
  pack_im2col2_kernel<<<cdiv(4104L * 2048, 256), blk, 0, stream>>>(Y1, Xc2);
  conv_gemm_kernel<<<514, blk, 0, stream>>>(cw2f, conv2_b, Xc2, Y2, 512, 2048, 513);

  // LSTM layer 0
  pack_x0_kernel<<<cdiv(4104L * 576, 256), blk, 0, stream>>>(Y2, subj_emb, subjects, Xl0);
  gemm_n_kernel<<<2056, blk, 0, stream>>>(wih0f, Xl0, b_ih0, b_hh0, G, 2048, 576);
  lstm_rec_kernel<<<1, 1024, 0, stream>>>(whh0f, G, H0f, H16_0, 513);
  // LSTM layer 1
  gemm_n_kernel<<<2056, blk, 0, stream>>>(wih1f, H16_0, b_ih1, b_hh1, G, 2048, 512);
  lstm_rec_kernel<<<1, 1024, 0, stream>>>(whh1f, G, H1f, H16_1, 513);

  // attention
  proj_gemm_kernel<<<514, blk, 0, stream>>>(qwf, q_b, H16_1, Qf, 512, 0);
  proj_gemm_kernel<<<514, blk, 0, stream>>>(kwf, k_b, H16_1, Kf, 512, 0);
  proj_gemm_kernel<<<514, blk, 0, stream>>>(cwf, c_b, H16_1, CNTf, 512, 1);
  qk_dots_kernel<<<dim3(137, 64), blk, 0, stream>>>(Qf, Kf, DOTS);
  prel_kernel<<<dim3(29, 64), blk, 0, stream>>>(Qf, relBf, P);
  softmax_kernel<<<4104, blk, 0, stream>>>(DOTS, P, WTf, WRf);
  av_gemm_kernel<<<dim3(17, 64), blk, 0, stream>>>(CNTf, WTf, ATT);
  relout_gemm_kernel<<<dim3(17, 64), blk, 0, stream>>>(WRf, relTf, ATT);
  pack_f16_kernel<<<cdiv(4104L * 512, 256), blk, 0, stream>>>(ATT, ATTf, 4104L * 512);
  gemm_n_kernel<<<514, blk, 0, stream>>>(fcwf, ATTf, fc_b, nullptr, FC, 512, 512);
  bn_reduce_kernel<<<64, blk, 0, stream>>>(FC, stats, 4104);
  bn_apply_kernel<<<8208, blk, 0, stream>>>(FC, H1f, stats, bn_g, bn_b,
                                            attn_scale, XF, 4104L * 512);
  // output projection
  pack_f16_kernel<<<cdiv(4104L * 512, 256), blk, 0, stream>>>(XF, XFf, 4104L * 512);
  proj_gemm_kernel<<<129, blk, 0, stream>>>(outwf, out_b, XFf, d_out, 128, 3);
}
